// Lstm_model_36326833390331
// MI455X (gfx1250) — matrix operand order repaired and checked by reading, not compiled, not run on hardware
//
#include <hip/hip_runtime.h>
#include <math.h>

// ---------------------------------------------------------------------------
// Types for CDNA5 WMMA (wave32): bf16 16x16x32 -> f32 accum
// ---------------------------------------------------------------------------
typedef __bf16 bf16_t;
typedef __attribute__((ext_vector_type(16))) __bf16 v16bf;
typedef __attribute__((ext_vector_type(8)))  __bf16 v8bf;
typedef __attribute__((ext_vector_type(8)))  float  v8f;

typedef int v4i __attribute__((ext_vector_type(4)));
typedef __attribute__((address_space(1))) v4i as1_v4i;   // global int4
typedef __attribute__((address_space(3))) v4i as3_v4i;   // LDS int4

#if defined(__AMDGCN__) && __has_builtin(__builtin_amdgcn_global_load_async_to_lds_b128)
#define HAVE_ASYNC_LDS 1
#else
#define HAVE_ASYNC_LDS 0
#endif

__device__ __forceinline__ void wait_async_zero() {
#if __has_builtin(__builtin_amdgcn_s_wait_asynccnt)
  __builtin_amdgcn_s_wait_asynccnt(0);
#elif defined(__AMDGCN__)
  asm volatile("s_wait_asynccnt 0" ::: "memory");
#endif
}

#define SEQ    2048
#define EMB    1024
#define HID    1024
#define G4     4096      // 4 * HID
#define OUTV   50257

__device__ __forceinline__ bf16_t f2bf(float f) {
  union { float f; unsigned u; } v; v.f = f;
  unsigned r = v.u + 0x7FFFu + ((v.u >> 16) & 1u);   // round-to-nearest-even
  unsigned short h = (unsigned short)(r >> 16);
  union { unsigned short s; bf16_t b; } o; o.s = h;
  return o.b;
}

__device__ __forceinline__ float sigmoidf_(float x) {
  return 1.0f / (1.0f + expf(-x));
}

// ---------------------------------------------------------------------------
// Init: combined gate bias, zero h double-buffer, zero grid-barrier counter
// ---------------------------------------------------------------------------
__global__ void init_kernel(const float* __restrict__ b_ih,
                            const float* __restrict__ b_hh,
                            float* __restrict__ bias_g,
                            float* __restrict__ hbuf,     // [2][HID]
                            int* __restrict__ bar) {
  int i = blockIdx.x * blockDim.x + threadIdx.x;
  if (i < G4)       bias_g[i] = b_ih[i] + b_hh[i];
  if (i < 2 * HID)  hbuf[i]   = 0.0f;
  if (i == 0)       *bar      = 0;
}

// ---------------------------------------------------------------------------
// Embedding gather + cast to bf16:  x_bf16[s][e] = bf16(emb[seq[s]][e])
// ---------------------------------------------------------------------------
__global__ __launch_bounds__(256) void gather_cast(const int*   __restrict__ idx,
                                                   const float* __restrict__ emb,
                                                   bf16_t*      __restrict__ dst) {
  const int s   = blockIdx.x;
  const int tok = idx[s];
  const float* src = emb + (size_t)tok * EMB;
  bf16_t*      d   = dst + (size_t)s   * EMB;
  for (int e = threadIdx.x; e < EMB; e += 256) d[e] = f2bf(src[e]);
}

// ---------------------------------------------------------------------------
// Bulk f32 -> bf16 cast (weights, hs)
// ---------------------------------------------------------------------------
__global__ __launch_bounds__(256) void cast_f32_bf16(const float* __restrict__ src,
                                                     bf16_t*      __restrict__ dst,
                                                     size_t n) {
  size_t i      = (size_t)blockIdx.x * blockDim.x + threadIdx.x;
  size_t stride = (size_t)gridDim.x * blockDim.x;
  for (; i < n; i += stride) dst[i] = f2bf(src[i]);
}

// ---------------------------------------------------------------------------
// Tiled WMMA GEMM:  C[M,N] = A[M,K] * W[N,K]^T + bias[N]
//   A, W row-major bf16 (both K-contiguous -> direct fragment loads)
//   Block tile 128x128, 8 waves (4 x 2), each wave 32x64 = 2x4 WMMA tiles.
//   K staged through LDS in steps of 32 via async direct-to-LDS loads
//   (ASYNCcnt path) when the toolchain exposes them.
// ---------------------------------------------------------------------------
#define LDS_PAD 40

__global__ __launch_bounds__(256) void gemm_bf16_wmma(
    const bf16_t* __restrict__ A,
    const bf16_t* __restrict__ W,
    const float*  __restrict__ bias,
    float*        __restrict__ C,
    int M, int N, int K)
{
  __shared__ bf16_t lA[128][LDS_PAD];
  __shared__ bf16_t lB[128][LDS_PAD];

  const int t    = threadIdx.x;
  const int lane = t & 31;
  const int wave = t >> 5;          // 0..7
  const int wm   = wave >> 1;       // 0..3  -> 32 rows of M each
  const int wn   = wave & 1;        // 0..1  -> 64 cols of N each
  const int lr   = lane & 15;
  const int hi   = lane >> 4;       // lane half (ISA 7.12.2 layouts)
  const int m0   = blockIdx.y * 128;
  const int n0   = blockIdx.x * 128;

  v8f acc[2][4];
#pragma unroll
  for (int i = 0; i < 2; ++i)
#pragma unroll
    for (int j = 0; j < 4; ++j)
#pragma unroll
      for (int r = 0; r < 8; ++r) acc[i][j][r] = 0.0f;

  for (int k0 = 0; k0 < K; k0 += 32) {
    // branch-free prefetch distance (re-prefetching the last tile is harmless)
    const int kp = (k0 + 32 < K) ? (k0 + 32) : k0;

    // ---- stage 128x32 A tile and 128x32 W tile into LDS (bf16) ----
#pragma unroll
    for (int p = 0; p < 2; ++p) {
      const int off = p * 2048 + t * 8;
      const int r   = off >> 5;
      const int kk  = off & 31;
      int nrow = n0 + r; if (nrow > N - 1) nrow = N - 1;   // clamp N tail
      const bf16_t* gA = &A[(size_t)(m0 + r) * K + k0 + kk];
      const bf16_t* gW = &W[(size_t)nrow     * K + k0 + kk];
#if HAVE_ASYNC_LDS
      // async global -> LDS (no VGPR round trip), tracked by ASYNCcnt.
      // Two-step casts: retype to generic v4i*, then addrspacecast.
      __builtin_amdgcn_global_load_async_to_lds_b128(
          (as1_v4i*)(v4i*)(void*)gA, (as3_v4i*)(v4i*)(void*)&lA[r][kk], 0, 0);
      __builtin_amdgcn_global_load_async_to_lds_b128(
          (as1_v4i*)(v4i*)(void*)gW, (as3_v4i*)(v4i*)(void*)&lB[r][kk], 0, 0);
#else
      *(v8bf*)&lA[r][kk] = *(const v8bf*)gA;
      *(v8bf*)&lB[r][kk] = *(const v8bf*)gW;
#endif
      __builtin_prefetch(&A[(size_t)(m0 + r) * K + kp + kk], 0, 0);
      __builtin_prefetch(&W[(size_t)nrow     * K + kp + kk], 0, 0);
    }
#if HAVE_ASYNC_LDS
    wait_async_zero();
#endif
    __syncthreads();

    // ---- A fragments (16x32): lane<16 holds K{0-7,16-23}, lane>=16 K{8-15,24-31}
    v16bf afrag[2];
#pragma unroll
    for (int i = 0; i < 2; ++i) {
      const int mr = wm * 32 + i * 16 + lr;
      union { v16bf v; v8bf h[2]; } u;
      u.h[0] = *(const v8bf*)&lA[mr][hi * 8];
      u.h[1] = *(const v8bf*)&lA[mr][16 + hi * 8];
      afrag[i] = u.v;
    }
    // ---- B fragments (32x16): lane<16 holds K{0-7,16-23}, lane>=16 K{8-15,24-31} of column lane%16
    v16bf bfrag[4];
#pragma unroll
    for (int j = 0; j < 4; ++j) {
      const int nr = wn * 64 + j * 16 + lr;
      union { v16bf v; v8bf h[2]; } u;
      u.h[0] = *(const v8bf*)&lB[nr][hi * 8];
      u.h[1] = *(const v8bf*)&lB[nr][16 + hi * 8];
      bfrag[j] = u.v;
    }

#pragma unroll
    for (int i = 0; i < 2; ++i)
#pragma unroll
      for (int j = 0; j < 4; ++j)
        acc[i][j] = __builtin_amdgcn_wmma_f32_16x16x32_bf16(
            false, afrag[i], false, bfrag[j], (short)0, acc[i][j], false, false);

    __syncthreads();
  }

  // ---- store: VGPR r -> M = r + hi*8 ; N = lane%16 ----
#pragma unroll
  for (int i = 0; i < 2; ++i)
#pragma unroll
    for (int j = 0; j < 4; ++j)
#pragma unroll
      for (int r = 0; r < 8; ++r) {
        const int gm = m0 + wm * 32 + i * 16 + hi * 8 + r;
        const int gn = n0 + wn * 64 + j * 16 + lr;
        if (gn < N) C[(size_t)gm * N + gn] = acc[i][j][r] + bias[gn];
      }
}

// ---------------------------------------------------------------------------
// Persistent LSTM scan. 16 WGs x 256 threads = 4096 threads = one gate row
// each. WG w owns hidden units [w*64, w*64+64); c stays in registers of
// threads 0..63. h double-buffered in global; one grid barrier per step.
// W_hh (bf16, 8 MB) stays L2-resident across all 2048 steps.
// ---------------------------------------------------------------------------
__device__ __forceinline__ void grid_barrier(int* cnt, int target) {
  __syncthreads();
  if (threadIdx.x == 0) {
    __hip_atomic_fetch_add(cnt, 1, __ATOMIC_ACQ_REL, __HIP_MEMORY_SCOPE_AGENT);
    while (__hip_atomic_load(cnt, __ATOMIC_ACQUIRE, __HIP_MEMORY_SCOPE_AGENT) < target) {
      __builtin_amdgcn_s_sleep(1);
    }
  }
  __syncthreads();
}

__global__ __launch_bounds__(256) void lstm_scan(
    const float*  __restrict__ xg,     // [SEQ][G4] precomputed x-gates (+bias)
    const bf16_t* __restrict__ Whh,    // [G4][HID] bf16
    float*        __restrict__ hbuf,   // [2][HID] double buffer
    float*        __restrict__ hs,     // [SEQ][HID] output
    int*          __restrict__ bar)
{
  __shared__ float lh[HID];
  __shared__ float gbuf[4][64];

  const int t    = threadIdx.x;
  const int w    = blockIdx.x;        // 0..15
  const int gate = t >> 6;            // 0..3 : PyTorch order i,f,g,o
  const int lu   = t & 63;
  const int row  = gate * HID + (w * 64 + lu);
  const bf16_t* wrow = Whh + (size_t)row * HID;
  const int nwg  = (int)gridDim.x;

  float creg = 0.0f;                  // cell state, owned by threads t<64
  int epoch = 0;

  for (int step = 0; step < SEQ; ++step) {
    const float* hcur  = hbuf + (step & 1) * HID;
    float*       hnext = hbuf + ((step & 1) ^ 1) * HID;

    // snapshot h into LDS
#pragma unroll
    for (int i = 0; i < 4; ++i) lh[t + i * 256] = hcur[t + i * 256];
    __syncthreads();

    // g[row] = xg[step][row] + dot(Whh[row], h)
    float acc = xg[(size_t)step * G4 + row];
#pragma unroll 4
    for (int k = 0; k < HID; k += 8) {
      const v8bf wv = *(const v8bf*)&wrow[k];
      acc += (float)wv[0] * lh[k + 0] + (float)wv[1] * lh[k + 1]
           + (float)wv[2] * lh[k + 2] + (float)wv[3] * lh[k + 3]
           + (float)wv[4] * lh[k + 4] + (float)wv[5] * lh[k + 5]
           + (float)wv[6] * lh[k + 6] + (float)wv[7] * lh[k + 7];
    }
    gbuf[gate][lu] = acc;
    __syncthreads();

    if (t < 64) {
      const float iv = sigmoidf_(gbuf[0][t]);
      const float fv = sigmoidf_(gbuf[1][t]);
      const float gv = tanhf   (gbuf[2][t]);
      const float ov = sigmoidf_(gbuf[3][t]);
      creg = fv * creg + iv * gv;
      const float hv = ov * tanhf(creg);
      const int u = w * 64 + t;
      hnext[u] = hv;
      hs[(size_t)step * HID + u] = hv;
    }
    ++epoch;
    grid_barrier(bar, epoch * nwg);
  }
}

// ---------------------------------------------------------------------------
// In-place row-wise log-softmax, one WG per row of length N
// ---------------------------------------------------------------------------
__global__ __launch_bounds__(256) void logsoftmax_rows(float* __restrict__ X, int N) {
  __shared__ float red[256];
  float* row = X + (size_t)blockIdx.x * N;

  float m = -INFINITY;
  for (int i = threadIdx.x; i < N; i += 256) m = fmaxf(m, row[i]);
  red[threadIdx.x] = m; __syncthreads();
  for (int s = 128; s > 0; s >>= 1) {
    if (threadIdx.x < s) red[threadIdx.x] = fmaxf(red[threadIdx.x], red[threadIdx.x + s]);
    __syncthreads();
  }
  m = red[0]; __syncthreads();

  float sum = 0.0f;
  for (int i = threadIdx.x; i < N; i += 256) sum += expf(row[i] - m);
  red[threadIdx.x] = sum; __syncthreads();
  for (int s = 128; s > 0; s >>= 1) {
    if (threadIdx.x < s) red[threadIdx.x] += red[threadIdx.x + s];
    __syncthreads();
  }
  const float lse = m + logf(red[0]);

  for (int i = threadIdx.x; i < N; i += 256) row[i] = row[i] - lse;
}

// ---------------------------------------------------------------------------
// Host launcher
// ---------------------------------------------------------------------------
extern "C" void kernel_launch(void* const* d_in, const int* in_sizes, int n_in,
                              void* d_out, int out_size, void* d_ws, size_t ws_size,
                              hipStream_t stream) {
  (void)in_sizes; (void)n_in; (void)out_size; (void)ws_size;

  const int*   input_seq = (const int*)  d_in[0];   // [SEQ]
  const float* emb       = (const float*)d_in[1];   // [VOCAB, EMB]
  const float* W_ih      = (const float*)d_in[2];   // [G4, EMB]
  const float* W_hh      = (const float*)d_in[3];   // [G4, HID]
  const float* b_ih      = (const float*)d_in[4];   // [G4]
  const float* b_hh      = (const float*)d_in[5];   // [G4]
  const float* W_out     = (const float*)d_in[6];   // [OUTV, HID]
  const float* b_out     = (const float*)d_in[7];   // [OUTV]
  float*       out       = (float*)d_out;           // [SEQ, OUTV]

  char*  ws  = (char*)d_ws;
  size_t off = 0;
  auto alloc = [&](size_t bytes) -> void* {
    void* p = ws + off;
    off += (bytes + 255) & ~(size_t)255;
    return p;
  };

  bf16_t* x_b    = (bf16_t*)alloc((size_t)SEQ  * EMB * 2);
  bf16_t* Wih_b  = (bf16_t*)alloc((size_t)G4   * EMB * 2);
  bf16_t* Whh_b  = (bf16_t*)alloc((size_t)G4   * HID * 2);
  bf16_t* Wout_b = (bf16_t*)alloc((size_t)OUTV * HID * 2);
  float*  xg     = (float*) alloc((size_t)SEQ  * G4  * 4);
  float*  hs     = (float*) alloc((size_t)SEQ  * HID * 4);
  bf16_t* hs_b   = (bf16_t*)alloc((size_t)SEQ  * HID * 2);
  float*  bias_g = (float*) alloc((size_t)G4 * 4);
  float*  hbuf   = (float*) alloc((size_t)2 * HID * 4);
  int*    bar    = (int*)   alloc(256);

  // 1) init: fused gate bias, zero h buffers, zero barrier counter
  init_kernel<<<16, 256, 0, stream>>>(b_ih, b_hh, bias_g, hbuf, bar);

  // 2) embedding gather -> bf16
  gather_cast<<<SEQ, 256, 0, stream>>>(input_seq, emb, x_b);

  // 3) weight casts -> bf16
  cast_f32_bf16<<<2048, 256, 0, stream>>>(W_ih,  Wih_b,  (size_t)G4 * EMB);
  cast_f32_bf16<<<2048, 256, 0, stream>>>(W_hh,  Whh_b,  (size_t)G4 * HID);
  cast_f32_bf16<<<8192, 256, 0, stream>>>(W_out, Wout_b, (size_t)OUTV * HID);

  // 4) x_gates = x * W_ih^T + (b_ih + b_hh)   [2048 x 4096]
  gemm_bf16_wmma<<<dim3(G4 / 128, SEQ / 128), 256, 0, stream>>>(
      x_b, Wih_b, bias_g, xg, SEQ, G4, EMB);

  // 5) persistent sequential LSTM scan (16 WGs, grid barrier per step)
  lstm_scan<<<16, 256, 0, stream>>>(xg, Whh_b, hbuf, hs, bar);

  // 6) hs -> bf16
  cast_f32_bf16<<<2048, 256, 0, stream>>>(hs, hs_b, (size_t)SEQ * HID);

  // 7) logits = hs * W_out^T + b_out   [2048 x 50257], N-tail guarded
  gemm_bf16_wmma<<<dim3((OUTV + 127) / 128, SEQ / 128), 256, 0, stream>>>(
      hs_b, Wout_b, b_out, out, SEQ, OUTV, HID);

  // 8) in-place log-softmax per row
  logsoftmax_rows<<<SEQ, 256, 0, stream>>>(out, OUTV);
}
